// GPT_41961830482332
// MI455X (gfx1250) — matrix operand order repaired and checked by reading, not compiled, not run on hardware
//
#include <hip/hip_runtime.h>
#include <math.h>

// ---------------- problem constants ----------------
constexpr int Bc = 2, Tc = 1024, Cc = 768, NHc = 12, Lc = 4, Vc = 50304;
constexpr int HSc = 64;           // head size
constexpr int Mc = Bc * Tc;       // 2048 rows of activations
constexpr int C3 = 3 * Cc;        // 2304

typedef __bf16 bf16;
typedef bf16  bf16x16 __attribute__((ext_vector_type(16)));
typedef float f32x8   __attribute__((ext_vector_type(8)));

// ---------------- fp32 -> bf16 convert (grid-stride) ----------------
__global__ void cvt_f32_to_bf16(const float* __restrict__ in, bf16* __restrict__ out, size_t n) {
    size_t i = (size_t)blockIdx.x * blockDim.x + threadIdx.x;
    size_t stride = (size_t)gridDim.x * blockDim.x;
    for (; i < n; i += stride) out[i] = (bf16)in[i];
}

// ---------------- embedding: x = wte[idx] + wpe ----------------
__global__ void embed_kernel(const int* __restrict__ idx, const float* __restrict__ wte,
                             const float* __restrict__ wpe,
                             float* __restrict__ xf, bf16* __restrict__ xh) {
    size_t i = (size_t)blockIdx.x * blockDim.x + threadIdx.x; // over Mc*Cc
    int c = (int)(i % Cc);
    int m = (int)(i / Cc);      // m = b*T + t
    int t = m % Tc;
    int tok = idx[m];
    float v = wte[(size_t)tok * Cc + c] + wpe[(size_t)t * Cc + c];
    xf[i] = v;
    xh[i] = (bf16)v;
}

// ---------------- WMMA bf16 GEMM: C[M,N] = A[M,K] * W[N,K]^T ----------------
// A fragment per ISA 7.12.2 (16-bit A 16x32): lane L (r=L&15, h=L>>4) holds
// row r, K = h*8+[0..7] in elems 0..7 and K = 16+h*8+[0..7] in elems 8..15.
__device__ __forceinline__ bf16x16 load_frag_a(const bf16* __restrict__ base, int ld) {
    int lane = threadIdx.x & 31;
    int r = lane & 15, h = lane >> 4;
    const bf16* p0 = base + (size_t)r * ld + h * 8;
    const bf16* p1 = base + (size_t)r * ld + 16 + h * 8;
    union { bf16x16 v; uint4 q[2]; } u;
    u.q[0] = *(const uint4*)p0;      // global_load_b128
    u.q[1] = *(const uint4*)p1;      // global_load_b128
    return u.v;
}
// B fragment (32x16, K x N): column n = lane&15 lives in lane; lane half h
// holds K = h*8+[0..7] in elems 0..7 and K = 16+h*8+[0..7] in elems 8..15.
// Since B = W^T with W row-major [N,K], column n of B is row n of W -> two 16B loads.
__device__ __forceinline__ bf16x16 load_frag_b(const bf16* __restrict__ base, int ld) {
    int lane = threadIdx.x & 31;
    int n = lane & 15, h = lane >> 4;
    const bf16* p = base + (size_t)n * ld + h * 8;
    union { bf16x16 v; uint4 q[2]; } u;
    u.q[0] = *(const uint4*)p;
    u.q[1] = *(const uint4*)(p + 16);
    return u.v;
}

// Block = 256 threads (8 waves), block tile 128(M) x 64(N); wave tile 32x32
// (2x2 WMMA accumulators). Optional residual add (resid may alias Cf) and
// optional bf16 mirror of the output (feeds the next GEMM).
__global__ void gemm_bf16_wmma(const bf16* __restrict__ A, const bf16* __restrict__ W,
                               float* __restrict__ Cf, bf16* __restrict__ Cb,
                               const float* __restrict__ resid,
                               int M, int N, int K) {
    const int w  = threadIdx.x >> 5;
    const int wm = blockIdx.y * 128 + (w & 3) * 32;   // row of this wave's tile
    const int wn = blockIdx.x * 64  + (w >> 2) * 32;  // col of this wave's tile

    f32x8 acc[2][2] = {};

    for (int kb = 0; kb < K; kb += 32) {
        if (kb + 32 < K) {  // speculative prefetch of next K-slab -> global_prefetch_b8
            __builtin_prefetch(A + (size_t)wm * K + kb + 32, 0, 3);
            __builtin_prefetch(W + (size_t)wn * K + kb + 32, 0, 3);
        }
        bf16x16 a0 = load_frag_a(A + (size_t)wm        * K + kb, K);
        bf16x16 a1 = load_frag_a(A + (size_t)(wm + 16) * K + kb, K);
        bf16x16 b0 = load_frag_b(W + (size_t)wn        * K + kb, K);
        bf16x16 b1 = load_frag_b(W + (size_t)(wn + 16) * K + kb, K);
        acc[0][0] = __builtin_amdgcn_wmma_f32_16x16x32_bf16(false, a0, false, b0, short(0), acc[0][0], false, false);
        acc[0][1] = __builtin_amdgcn_wmma_f32_16x16x32_bf16(false, a0, false, b1, short(0), acc[0][1], false, false);
        acc[1][0] = __builtin_amdgcn_wmma_f32_16x16x32_bf16(false, a1, false, b0, short(0), acc[1][0], false, false);
        acc[1][1] = __builtin_amdgcn_wmma_f32_16x16x32_bf16(false, a1, false, b1, short(0), acc[1][1], false, false);
    }

    // D layout (32-bit C/D 16x16): lane L -> col = wn + (L&15); VGPR r -> row
    // (L>>4)*8 + r within the tile.
    const int lane = threadIdx.x & 31;
    const int nn = lane & 15, hh = lane >> 4;
    #pragma unroll
    for (int i = 0; i < 2; ++i) {
        #pragma unroll
        for (int j = 0; j < 2; ++j) {
            int col = wn + j * 16 + nn;
            #pragma unroll
            for (int r = 0; r < 8; ++r) {
                int row = wm + i * 16 + hh * 8 + r;
                size_t off = (size_t)row * N + col;
                float v = acc[i][j][r];
                if (resid) v += resid[off];
                Cf[off] = v;
                if (Cb) Cb[off] = (bf16)v;
            }
        }
    }
}

// ---------------- per-(b,t,s,h) norm over HS + erf + rwkv prep ----------------
// one block of 64 threads per group; gid = ((b*T + t)*3 + s)*NH + h
__global__ void prep_qkv_kernel(const float* __restrict__ qkv, const float* __restrict__ eta_l,
                                float* __restrict__ qA, float* __restrict__ aA,
                                float* __restrict__ keA, float* __restrict__ vA) {
    __shared__ float red[HSc];
    int gid = blockIdx.x;
    int i = threadIdx.x;
    int h = gid % NHc;
    int s = (gid / NHc) % 3;
    int t = (gid / (3 * NHc)) % Tc;
    int b = gid / (3 * NHc * Tc);

    float val = qkv[((size_t)(b * Tc + t)) * C3 + s * Cc + h * HSc + i];

    red[i] = val; __syncthreads();
    #pragma unroll
    for (int off = 32; off > 0; off >>= 1) { if (i < off) red[i] += red[i + off]; __syncthreads(); }
    float mu = red[0] / (float)HSc;
    __syncthreads();
    float d = val - mu;
    red[i] = d * d; __syncthreads();
    #pragma unroll
    for (int off = 32; off > 0; off >>= 1) { if (i < off) red[i] += red[i + off]; __syncthreads(); }
    float sd = sqrtf(red[0] / (float)(HSc - 1));   // ddof=1
    float xn = d / sd;

    size_t o = (((size_t)(b * NHc + h) * Tc + t) * HSc + i);
    if (s == 0) {
        qA[o] = erff(xn);
    } else if (s == 1) {
        float k = erff(xn);
        aA[o]  = -k;
        keA[o] = k * eta_l[h * HSc + i];   // b_t == k_t == k * eta
    } else {
        vA[o] = xn;
    }
}

// ---------------- sequential RWKV7 scan: one block per (b, head) ----------------
// 64 threads; thread i owns state row i (64 floats in registers).
// s[i][j] = s[i][j]*w[i] + (dot(s[i],a) + v[i]) * ke[j];  out[i] = dot(q, s[i])
__global__ void rwkv_scan_kernel(const float* __restrict__ qA, const float* __restrict__ aA,
                                 const float* __restrict__ keA, const float* __restrict__ vA,
                                 const float* __restrict__ wdecay_l, float* __restrict__ yf) {
    __shared__ float aS[HSc], keS[HSc], qS[HSc];
    int bh = blockIdx.x;
    int b = bh / NHc, h = bh % NHc;
    int i = threadIdx.x;
    float w_i = wdecay_l[h * HSc + i];

    float s[HSc];
    #pragma unroll
    for (int j = 0; j < HSc; ++j) s[j] = 0.f;

    size_t base = ((size_t)(b * NHc + h) * Tc) * HSc;
    for (int t = 0; t < Tc; ++t) {
        size_t o = base + (size_t)t * HSc + i;
        aS[i]  = aA[o];
        keS[i] = keA[o];
        qS[i]  = qA[o];
        float v_i = vA[o];
        __syncthreads();

        float dot = 0.f;
        #pragma unroll
        for (int k = 0; k < HSc; ++k) dot += s[k] * aS[k];
        float coef = dot + v_i;

        float out = 0.f;
        #pragma unroll
        for (int j = 0; j < HSc; ++j) {
            s[j] = s[j] * w_i + coef * keS[j];
            out += qS[j] * s[j];
        }
        yf[((size_t)b * Tc + t) * Cc + h * HSc + i] = out;
        __syncthreads();
    }
}

// ---------------- final norm of the last position (ddof=1 over C) ----------------
__global__ void final_norm_kernel(const float* __restrict__ xf, float* __restrict__ xn) {
    __shared__ float red[256];
    int b = blockIdx.x;
    int tid = threadIdx.x;
    const float* row = xf + ((size_t)b * Tc + (Tc - 1)) * Cc;
    float v0 = row[tid], v1 = row[tid + 256], v2 = row[tid + 512];

    red[tid] = v0 + v1 + v2; __syncthreads();
    for (int off = 128; off > 0; off >>= 1) { if (tid < off) red[tid] += red[tid + off]; __syncthreads(); }
    float mu = red[0] / (float)Cc;
    __syncthreads();
    float d0 = v0 - mu, d1 = v1 - mu, d2 = v2 - mu;
    red[tid] = d0 * d0 + d1 * d1 + d2 * d2; __syncthreads();
    for (int off = 128; off > 0; off >>= 1) { if (tid < off) red[tid] += red[tid + off]; __syncthreads(); }
    float sd = sqrtf(red[0] / (float)(Cc - 1));
    xn[(size_t)b * Cc + tid]       = d0 / sd;
    xn[(size_t)b * Cc + tid + 256] = d1 / sd;
    xn[(size_t)b * Cc + tid + 512] = d2 / sd;
}

// ---------------- logits GEMV: one wave32 per (b, v); bandwidth bound ----------------
__global__ void logits_kernel(const float* __restrict__ xn, const float* __restrict__ lmw,
                              float* __restrict__ out) {
    int wid = blockIdx.x * (blockDim.x >> 5) + (threadIdx.x >> 5);
    int lane = threadIdx.x & 31;
    if (wid >= Bc * Vc) return;
    int b = wid / Vc, v = wid % Vc;
    const float* wrow = lmw + (size_t)v * Cc;
    const float* xrow = xn + (size_t)b * Cc;
    float acc = 0.f;
    #pragma unroll 4
    for (int c = lane; c < Cc; c += 32) acc += xrow[c] * wrow[c];
    #pragma unroll
    for (int off = 16; off > 0; off >>= 1) acc += __shfl_xor(acc, off, 32);
    if (lane == 0) out[(size_t)b * Vc + v] = acc;
}

// ---------------- host orchestration ----------------
extern "C" void kernel_launch(void* const* d_in, const int* in_sizes, int n_in,
                              void* d_out, int out_size, void* d_ws, size_t ws_size,
                              hipStream_t stream) {
    (void)in_sizes; (void)n_in; (void)out_size; (void)ws_size;
    const int*   idx     = (const int*)d_in[0];
    const float* wte     = (const float*)d_in[1];
    const float* wpe     = (const float*)d_in[2];
    const float* c_attn  = (const float*)d_in[3];   // [L, 3C, C]
    const float* c_proj  = (const float*)d_in[4];   // [L, C, C]
    const float* w_decay = (const float*)d_in[5];   // [L, C]
    const float* eta     = (const float*)d_in[6];   // [L, C]
    const float* lmw     = (const float*)d_in[7];   // [V, C]
    float* logits = (float*)d_out;                  // [B, 1, V] fp32

    char* ws = (char*)d_ws;
    size_t o = 0;
    auto carve = [&](size_t bytes) { char* p = ws + o; o += (bytes + 255) & ~(size_t)255; return p; };

    float* xf    = (float*)carve((size_t)Mc * Cc * 4);     // activations fp32
    bf16*  xh    = (bf16*) carve((size_t)Mc * Cc * 2);     // activations bf16
    float* qkvf  = (float*)carve((size_t)Mc * C3 * 4);     // QKV output
    float* qA    = (float*)carve((size_t)Mc * Cc * 4);
    float* aA    = (float*)carve((size_t)Mc * Cc * 4);
    float* keA   = (float*)carve((size_t)Mc * Cc * 4);
    float* vA    = (float*)carve((size_t)Mc * Cc * 4);
    bf16*  wqkvh = (bf16*) carve((size_t)Lc * C3 * Cc * 2);
    bf16*  wprjh = (bf16*) carve((size_t)Lc * Cc * Cc * 2);
    float* xn    = (float*)carve((size_t)Bc * Cc * 4);
    // y reuses the (dead-by-then) qkvf region
    float* yf = qkvf;
    bf16*  yh = (bf16*)((char*)qkvf + (size_t)Mc * Cc * 4);

    // 1) weights -> bf16
    cvt_f32_to_bf16<<<4096, 256, 0, stream>>>(c_attn, wqkvh, (size_t)Lc * C3 * Cc);
    cvt_f32_to_bf16<<<4096, 256, 0, stream>>>(c_proj, wprjh, (size_t)Lc * Cc * Cc);

    // 2) embedding
    embed_kernel<<<(Mc * Cc) / 256, 256, 0, stream>>>(idx, wte, wpe, xf, xh);

    // 3) layers
    for (int l = 0; l < Lc; ++l) {
        dim3 gq(C3 / 64, Mc / 128);
        gemm_bf16_wmma<<<gq, 256, 0, stream>>>(xh, wqkvh + (size_t)l * C3 * Cc,
                                               qkvf, (bf16*)nullptr, (const float*)nullptr,
                                               Mc, C3, Cc);
        prep_qkv_kernel<<<Bc * Tc * 3 * NHc, HSc, 0, stream>>>(qkvf, eta + (size_t)l * Cc,
                                                               qA, aA, keA, vA);
        rwkv_scan_kernel<<<Bc * NHc, HSc, 0, stream>>>(qA, aA, keA, vA,
                                                       w_decay + (size_t)l * Cc, yf);
        cvt_f32_to_bf16<<<2048, 256, 0, stream>>>(yf, yh, (size_t)Mc * Cc);
        dim3 gp(Cc / 64, Mc / 128);
        gemm_bf16_wmma<<<gp, 256, 0, stream>>>(yh, wprjh + (size_t)l * Cc * Cc,
                                               xf, xh, xf,          // residual in-place, bf16 mirror
                                               Mc, Cc, Cc);
    }

    // 4) final norm (last position only) + logits
    final_norm_kernel<<<Bc, 256, 0, stream>>>(xf, xn);
    logits_kernel<<<(Bc * Vc + 7) / 8, 256, 0, stream>>>(xn, lmw, logits);
}
